// MultiHeadAttention_38233798869216
// MI455X (gfx1250) — compile-verified, matrix operand order checked by reading, not run on hardware
//
#include <hip/hip_runtime.h>

// ---------------------------------------------------------------------------
// MHA forward for gfx1250 (MI455X): bf16 WMMA, f32 accumulate, flash-attention
// with async-to-LDS double-buffered K/V staging.
// B=4, S=2048, D_MODEL=1024, HEADS=16, D_K=64.
// ---------------------------------------------------------------------------

typedef __attribute__((ext_vector_type(16))) __bf16 v16bf;
typedef __attribute__((ext_vector_type(8)))  __bf16 v8bf;
typedef __attribute__((ext_vector_type(8)))  float  v8f;
typedef int v4i_ __attribute__((vector_size(16)));   // builtin's pointee type

static constexpr int S_LEN  = 2048;
static constexpr int DMODEL = 1024;
static constexpr int NHEAD  = 16;
static constexpr int DK     = 64;
static constexpr int BATCH  = 4;
static constexpr int MROWS  = BATCH * S_LEN;   // 8192

#if defined(__has_builtin)
#if __has_builtin(__builtin_amdgcn_global_load_async_to_lds_b128)
#define HAVE_ASYNC_LDS 1
#endif
#endif
#ifndef HAVE_ASYNC_LDS
#define HAVE_ASYNC_LDS 0
#endif

__device__ __forceinline__ __bf16 f2bf(float f) {
  unsigned u = __builtin_bit_cast(unsigned, f);
  unsigned r = u + 0x7FFFu + ((u >> 16) & 1u);   // round to nearest even
  unsigned short h = (unsigned short)(r >> 16);
  return __builtin_bit_cast(__bf16, h);
}

__device__ __forceinline__ v8f wmma_bf16(v16bf a, v16bf b, v8f c) {
  // v_wmma_f32_16x16x32_bf16: D = A(16x32) * B(32x16) + C(16x16 f32)
  return __builtin_amdgcn_wmma_f32_16x16x32_bf16(
      false, a, false, b, (short)0, c, false, false);
}

// 16x32 bf16 A-fragment loader (also serves B via B^T, which is row-major for
// every operand here). Lanes 0-15: row (lane&15), K = {0..7, 16..23};
// lanes 16-31: K = {8..15, 24..31}.
__device__ __forceinline__ v16bf load_a_frag(const __bf16* p, int ld, int lane) {
  int row = lane & 15;
  int ko  = (lane >> 4) << 3;
  const __bf16* base = p + (size_t)row * ld + ko;
  union { v16bf v; v8bf h[2]; } u;
  u.h[0] = *(const v8bf*)(base);
  u.h[1] = *(const v8bf*)(base + 16);
  return u.v;
}

// 16-byte global -> LDS copy: async (ASYNCcnt-tracked) when available.
__device__ __forceinline__ void copy16_g2l(const __bf16* g, __bf16* l) {
#if HAVE_ASYNC_LDS
  void* gp = (void*)g;   // drop const
  void* lp = (void*)l;
  __builtin_amdgcn_global_load_async_to_lds_b128(
      (__attribute__((address_space(1))) v4i_*)gp,
      (__attribute__((address_space(3))) v4i_*)lp, 0, 0);
#else
  *(int4*)l = *(const int4*)g;
#endif
}

__device__ __forceinline__ void async_wait0() {
#if HAVE_ASYNC_LDS
#if __has_builtin(__builtin_amdgcn_s_wait_asynccnt)
  __builtin_amdgcn_s_wait_asynccnt(0);
#else
  asm volatile("s_wait_asynccnt 0" ::: "memory");
#endif
#endif
}

// ---------------------------------------------------------------------------
// f32 -> bf16 conversion
// ---------------------------------------------------------------------------
__global__ void cvt_f32_bf16(const float* __restrict__ src,
                             __bf16* __restrict__ dst, int n) {
  int i = blockIdx.x * blockDim.x + threadIdx.x;
  if (i < n) dst[i] = f2bf(src[i]);
}

// ---------------------------------------------------------------------------
// GEMM: Y[M=8192, N=1024] = X[M,K=1024] @ W[N,K]^T    (bf16 in, f32 acc)
// 64x64 per-wave tile: 16 accumulators, 16 b128 loads per 16 WMMAs.
// MODE 0: bf16 -> [B,H,S,DK] (Q/K layout), scaled
// MODE 1: bf16 -> [B,H,DK,S] (V transposed layout)
// MODE 2: f32  -> [M,N]      (final output)
// ---------------------------------------------------------------------------
template <int MODE>
__device__ __forceinline__ void store_frag(void* outp, v8f c, int rBase,
                                           int cBase, int lane, float scale) {
  int col = cBase + (lane & 15);
  int rh  = (lane >> 4) << 3;
#pragma unroll
  for (int j = 0; j < 8; ++j) {
    int row = rBase + j + rh;
    float v = c[j] * scale;
    if (MODE == 2) {
      ((float*)outp)[(size_t)row * DMODEL + col] = v;
    } else {
      int b = row >> 11, s = row & (S_LEN - 1);
      int h = col >> 6, d = col & (DK - 1);
      size_t idx;
      if (MODE == 0)
        idx = ((size_t)(b * NHEAD + h) * S_LEN + s) * DK + d;
      else
        idx = ((size_t)(b * NHEAD + h) * DK + d) * S_LEN + s;
      ((__bf16*)outp)[idx] = f2bf(v);
    }
  }
}

template <int MODE>
__global__ __launch_bounds__(256) void gemm_xwT(const __bf16* __restrict__ X,
                                                const __bf16* __restrict__ W,
                                                void* __restrict__ outp,
                                                float scale) {
  int lane = threadIdx.x & 31;
  int wid  = threadIdx.x >> 5;
  int gw   = blockIdx.x * 8 + wid;       // (8192/64)*(1024/64) = 2048 waves
  int tm   = (gw >> 4) << 6;             // 128 M-tiles of 64
  int tn   = (gw & 15) << 6;             // 16  N-tiles of 64

  v8f acc[4][4];
#pragma unroll
  for (int i = 0; i < 4; ++i)
#pragma unroll
    for (int j = 0; j < 4; ++j) acc[i][j] = v8f{};

  for (int k = 0; k < DMODEL; k += 32) {
    v16bf a[4], b[4];
#pragma unroll
    for (int i = 0; i < 4; ++i)
      a[i] = load_a_frag(X + (size_t)(tm + 16 * i) * DMODEL + k, DMODEL, lane);
#pragma unroll
    for (int j = 0; j < 4; ++j)
      b[j] = load_a_frag(W + (size_t)(tn + 16 * j) * DMODEL + k, DMODEL, lane);
#pragma unroll
    for (int i = 0; i < 4; ++i)
#pragma unroll
      for (int j = 0; j < 4; ++j)
        acc[i][j] = wmma_bf16(a[i], b[j], acc[i][j]);
  }
#pragma unroll
  for (int i = 0; i < 4; ++i)
#pragma unroll
    for (int j = 0; j < 4; ++j)
      store_frag<MODE>(outp, acc[i][j], tm + 16 * i, tn + 16 * j, lane, scale);
}

// ---------------------------------------------------------------------------
// Flash attention (causal). One block = one (b,h) and 128 query rows
// (8 waves x 16 rows). All waves share causal key tiles of 32, staged into
// LDS double-buffered via async global->LDS copies.
// Q,K: [B,H,S,DK] bf16 (Q pre-scaled by 1/8). Vt: [B,H,DK,S] bf16.
// Output: bf16 [B,S,DMODEL] (heads re-interleaved for the final GEMM).
// ---------------------------------------------------------------------------
__global__ __launch_bounds__(256) void attn_fwd(const __bf16* __restrict__ Q,
                                                const __bf16* __restrict__ K,
                                                const __bf16* __restrict__ Vt,
                                                __bf16* __restrict__ Oout) {
  __shared__ __align__(16) __bf16 ldsK[2][32 * DK];   // 2 x 4KB, keys x dims
  __shared__ __align__(16) __bf16 ldsV[2][DK * 32];   // 2 x 4KB, dims x keys
  __shared__ __align__(16) __bf16 ldsP[8][16 * 32];   // per-wave P tiles

  int tid  = threadIdx.x;
  int lane = tid & 31;
  int wid  = tid >> 5;
  int bh   = blockIdx.x >> 4;            // 0..63
  int qc   = blockIdx.x & 15;            // 16 chunks of 128 query rows
  int qBase0 = qc * 128;
  int qBase  = qBase0 + wid * 16;

  const __bf16* Qb = Q  + (size_t)bh * S_LEN * DK;
  const __bf16* Kb = K  + (size_t)bh * S_LEN * DK;
  const __bf16* Vb = Vt + (size_t)bh * DK * S_LEN;

  v16bf qa0 = load_a_frag(Qb + (size_t)qBase * DK,      DK, lane);
  v16bf qa1 = load_a_frag(Qb + (size_t)qBase * DK + 32, DK, lane);

  v8f o0 = {}, o1 = {}, o2 = {}, o3 = {};
  float m[8], l[8];
#pragma unroll
  for (int j = 0; j < 8; ++j) { m[j] = -3.0e38f; l[j] = 0.0f; }

  int rh = (lane >> 4) << 3;
  // Uniform trip count for the whole block (keys 0 .. qBase0+127); waves with
  // smaller qBase see fully-masked tail tiles, which contribute p == 0.
  int nT = (qBase0 + 128 + 31) >> 5;

  // Cooperative stage of key tile kBase into buffer `buf`.
  auto stage = [&](int buf, int kBase) {
    {   // K tile: 32 rows x 64 dims = 4KB; 256 threads x 16B
      int row = tid >> 3, ch = (tid & 7) << 3;
      copy16_g2l(Kb + (size_t)(kBase + row) * DK + ch, &ldsK[buf][row * DK + ch]);
    }
    {   // V tile: 64 dims x 32 keys = 4KB; 256 threads x 16B
      int row = tid >> 2, ch = (tid & 3) << 3;
      copy16_g2l(Vb + (size_t)row * S_LEN + kBase + ch, &ldsV[buf][row * 32 + ch]);
    }
  };

  stage(0, 0);
  async_wait0();
  __syncthreads();

  for (int t = 0; t < nT; ++t) {
    int buf   = t & 1;
    int kBase = t * 32;
    if (t + 1 < nT) stage(buf ^ 1, (t + 1) * 32);   // overlap next fetch

    // ---- scores S(16q x 32k) = Q(16x64) @ K_tile^T (from LDS) ----
    v16bf kb0a = load_a_frag(&ldsK[buf][0],            DK, lane);
    v16bf kb0b = load_a_frag(&ldsK[buf][32],           DK, lane);
    v16bf kb1a = load_a_frag(&ldsK[buf][16 * DK],      DK, lane);
    v16bf kb1b = load_a_frag(&ldsK[buf][16 * DK + 32], DK, lane);
    v8f s0 = {}, s1 = {};
    s0 = wmma_bf16(qa0, kb0a, s0);
    s0 = wmma_bf16(qa1, kb0b, s0);
    s1 = wmma_bf16(qa0, kb1a, s1);
    s1 = wmma_bf16(qa1, kb1b, s1);

    // ---- causal mask (covers partially- and fully-masked tiles) ----
    if (kBase + 31 > qBase) {
      int col0 = kBase + (lane & 15);
#pragma unroll
      for (int j = 0; j < 8; ++j) {
        int row = qBase + j + rh;
        if (col0 > row)      s0[j] = -1.0e30f;
        if (col0 + 16 > row) s1[j] = -1.0e30f;
      }
    }

    // ---- online softmax update ----
#pragma unroll
    for (int j = 0; j < 8; ++j) {
      float rm = fmaxf(s0[j], s1[j]);
      rm = fmaxf(rm, __shfl_xor(rm, 1, 16));
      rm = fmaxf(rm, __shfl_xor(rm, 2, 16));
      rm = fmaxf(rm, __shfl_xor(rm, 4, 16));
      rm = fmaxf(rm, __shfl_xor(rm, 8, 16));
      float mn   = fmaxf(m[j], rm);
      float corr = __expf(m[j] - mn);
      float p0   = __expf(s0[j] - mn);
      float p1   = __expf(s1[j] - mn);
      float rs = p0 + p1;
      rs += __shfl_xor(rs, 1, 16);
      rs += __shfl_xor(rs, 2, 16);
      rs += __shfl_xor(rs, 4, 16);
      rs += __shfl_xor(rs, 8, 16);
      l[j] = l[j] * corr + rs;
      m[j] = mn;
      o0[j] *= corr; o1[j] *= corr; o2[j] *= corr; o3[j] *= corr;
      // stash P row-major [16 x 32] for the C-layout -> A-layout transpose
      int r = j + rh;
      ldsP[wid][r * 32 + (lane & 15)]      = f2bf(p0);
      ldsP[wid][r * 32 + 16 + (lane & 15)] = f2bf(p1);
    }

    // ---- O += P(16x32) @ V_tile(32x64), V^T fragments from LDS ----
    v16bf pf  = load_a_frag(&ldsP[wid][0], 32, lane);
    v16bf vb0 = load_a_frag(&ldsV[buf][0],       32, lane);
    v16bf vb1 = load_a_frag(&ldsV[buf][16 * 32], 32, lane);
    v16bf vb2 = load_a_frag(&ldsV[buf][32 * 32], 32, lane);
    v16bf vb3 = load_a_frag(&ldsV[buf][48 * 32], 32, lane);
    o0 = wmma_bf16(pf, vb0, o0);
    o1 = wmma_bf16(pf, vb1, o1);
    o2 = wmma_bf16(pf, vb2, o2);
    o3 = wmma_bf16(pf, vb3, o3);

    async_wait0();      // next tile's async copies complete
    __syncthreads();    // everyone done reading `buf`; next tile visible
  }

  // ---- normalize and write [B, S, DMODEL] with head interleave ----
  int b = bh >> 4, h = bh & 15;
  int col = lane & 15;
#pragma unroll
  for (int j = 0; j < 8; ++j) {
    float inv = 1.0f / l[j];
    int row = qBase + j + rh;
    size_t base = ((size_t)(b * S_LEN + row)) * DMODEL + h * DK + col;
    Oout[base +  0] = f2bf(o0[j] * inv);
    Oout[base + 16] = f2bf(o1[j] * inv);
    Oout[base + 32] = f2bf(o2[j] * inv);
    Oout[base + 48] = f2bf(o3[j] * inv);
  }
}

// ---------------------------------------------------------------------------
// Host-side orchestration. Inputs: q, mask(unused: causal computed in-kernel),
// w_q, w_k, w_v, w_o. Output: f32 [B, S, DMODEL]. Workspace: ~88 MB bf16.
// ---------------------------------------------------------------------------
extern "C" void kernel_launch(void* const* d_in, const int* in_sizes, int n_in,
                              void* d_out, int out_size, void* d_ws, size_t ws_size,
                              hipStream_t stream) {
  const float* q   = (const float*)d_in[0];
  const float* w_q = (const float*)d_in[2];
  const float* w_k = (const float*)d_in[3];
  const float* w_v = (const float*)d_in[4];
  const float* w_o = (const float*)d_in[5];

  const size_t nAct = (size_t)MROWS * DMODEL;     // 8,388,608
  const size_t nW   = (size_t)DMODEL * DMODEL;    // 1,048,576

  __bf16* qb  = (__bf16*)d_ws;
  __bf16* wqb = qb  + nAct;
  __bf16* wkb = wqb + nW;
  __bf16* wvb = wkb + nW;
  __bf16* wob = wvb + nW;
  __bf16* Qb  = wob + nW;     // [B,H,S,DK], pre-scaled by 1/sqrt(DK)
  __bf16* Kb  = Qb  + nAct;   // [B,H,S,DK]
  __bf16* Vtb = Kb  + nAct;   // [B,H,DK,S]
  __bf16* aOb = Vtb + nAct;   // [B,S,DMODEL]

  cvt_f32_bf16<<<(int)((nAct + 255) / 256), 256, 0, stream>>>(q,   qb,  (int)nAct);
  cvt_f32_bf16<<<(int)((nW   + 255) / 256), 256, 0, stream>>>(w_q, wqb, (int)nW);
  cvt_f32_bf16<<<(int)((nW   + 255) / 256), 256, 0, stream>>>(w_k, wkb, (int)nW);
  cvt_f32_bf16<<<(int)((nW   + 255) / 256), 256, 0, stream>>>(w_v, wvb, (int)nW);
  cvt_f32_bf16<<<(int)((nW   + 255) / 256), 256, 0, stream>>>(w_o, wob, (int)nW);

  // (M/64)*(N/64) = 128*16 = 2048 wave tiles / 8 waves per block = 256 blocks
  gemm_xwT<0><<<256, 256, 0, stream>>>(qb, wqb, (void*)Qb,  0.125f);
  gemm_xwT<0><<<256, 256, 0, stream>>>(qb, wkb, (void*)Kb,  1.0f);
  gemm_xwT<1><<<256, 256, 0, stream>>>(qb, wvb, (void*)Vtb, 1.0f);

  // B*H * (S/128) = 64*16 = 1024 blocks, 8 waves each
  attn_fwd<<<1024, 256, 0, stream>>>(Qb, Kb, Vtb, aOb);

  gemm_xwT<2><<<256, 256, 0, stream>>>(aOb, wob, d_out, 1.0f);
}
